// FusedMoE_33414845563703
// MI455X (gfx1250) — matrix operand order repaired and checked by reading, not compiled, not run on hardware
//
#include <hip/hip_runtime.h>
#include <hip/hip_bf16.h>
#include <math.h>

// ---------------- problem constants (match reference) ----------------
#define T_TOK 1024
#define HID   1024
#define NEXP  8
#define INTER 3584
#define TOPK  2
#define SROWS (T_TOK*TOPK + NEXP*16)   // 2176 expert-sorted rows (padded)
#define MTILES (SROWS/16)              // 136

typedef __bf16 bf16_t;
typedef __attribute__((ext_vector_type(16))) __bf16 v16bf;
typedef __attribute__((ext_vector_type(8)))  __bf16 v8bf;
typedef __attribute__((ext_vector_type(4)))  __bf16 v4bf;
typedef __attribute__((ext_vector_type(8)))  float  v8f;
typedef __attribute__((ext_vector_type(4)))  float  v4f;

// ---------------- helpers: WMMA fragment loads ----------------
// A (16x32 bf16, M=lane&15): lanes<16 hold K 0-7 & 16-23; lanes>=16 hold K 8-15 & 24-31.
__device__ __forceinline__ v16bf load_a_bf16(const bf16_t* row_ptr, int hi) {
    v8bf lo8 = *(const v8bf*)(row_ptr + hi * 8);
    v8bf hi8 = *(const v8bf*)(row_ptr + 16 + hi * 8);
    v16bf a;
#pragma unroll
    for (int i = 0; i < 8; ++i) { a[i] = lo8[i]; a[i + 8] = hi8[i]; }
    return a;
}

// B (32x16 bf16, N=lane&15): lanes<16 hold K 0-7 & 16-23; lanes>=16 hold K 8-15 & 24-31.
// Source is fp32 weights, converted in-register (saves a full weight-copy pass;
// v_cvt_pk_bf16_f32 co-executes with the XDL WMMA pipe).
__device__ __forceinline__ v16bf load_b_f32(const float* p) {
    v16bf b;
#pragma unroll
    for (int c = 0; c < 4; ++c) {
        v4f f = *(const v4f*)(p + (c >> 1) * 8 + c * 4);
#pragma unroll
        for (int j = 0; j < 4; ++j) b[c * 4 + j] = (bf16_t)f[j];
    }
    return b;
}

// fast sigmoid: v_exp_f32 + v_rcp_f32 (1 ulp) instead of the precise-div sequence
__device__ __forceinline__ float fast_silu(float g) {
    return g * __builtin_amdgcn_rcpf(1.f + __expf(-g));
}

// ---------------- 1: init scratch state ----------------
__global__ void moe_init(int* counts, int* cursors, int* token_of_row, float* row_weight) {
    int i = blockIdx.x * blockDim.x + threadIdx.x;
    if (i < NEXP) { counts[i] = 0; cursors[i] = 0; }
    if (i < SROWS) { token_of_row[i] = -1; row_weight[i] = 0.f; }
}

// ---------------- 2: routing (softmax + top-2 + renormalize) ----------------
__global__ void moe_route(const float* __restrict__ logits, int* counts,
                          int* __restrict__ topi, float* __restrict__ topw) {
    int t = blockIdx.x * blockDim.x + threadIdx.x;
    if (t >= T_TOK) return;
    float s[NEXP];
    float m = -1e30f;
#pragma unroll
    for (int e = 0; e < NEXP; ++e) { s[e] = logits[t * NEXP + e]; m = fmaxf(m, s[e]); }
#pragma unroll
    for (int e = 0; e < NEXP; ++e) s[e] = __expf(s[e] - m);   // softmax numerators (denom cancels)
    int i0 = 0;
#pragma unroll
    for (int e = 1; e < NEXP; ++e) if (s[e] > s[i0]) i0 = e;  // first-max tie-break like jax
    int i1 = (i0 == 0) ? 1 : 0;
#pragma unroll
    for (int e = 0; e < NEXP; ++e) if (e != i0 && s[e] > s[i1]) i1 = e;
    float inv = __builtin_amdgcn_rcpf(s[i0] + s[i1]);
    topi[2 * t] = i0; topi[2 * t + 1] = i1;
    topw[2 * t] = s[i0] * inv; topw[2 * t + 1] = s[i1] * inv;
    atomicAdd(&counts[i0], 1);
    atomicAdd(&counts[i1], 1);
}

// ---------------- 3: padded prefix offsets + tile->expert map ----------------
__global__ void moe_offsets(const int* __restrict__ counts, int* __restrict__ offs,
                            int* __restrict__ tile_expert) {
    __shared__ int so[NEXP + 1];
    int tid = threadIdx.x;
    if (tid == 0) {
        int acc = 0;
        for (int e = 0; e < NEXP; ++e) { so[e] = acc; acc += (counts[e] + 15) & ~15; }
        so[NEXP] = acc;
    }
    __syncthreads();
    if (tid <= NEXP) offs[tid] = so[tid];
    if (tid < MTILES) {
        int r = tid * 16, e = 0;
        for (int j = 0; j < NEXP; ++j) if (r >= so[j] && r < so[j + 1]) e = j;
        tile_expert[tid] = e;
    }
}

// ---------------- 4: scatter tokens into expert-sorted rows ----------------
__global__ void moe_scatter(const int* __restrict__ topi, const float* __restrict__ topw,
                            const int* __restrict__ offs, int* cursors,
                            int* __restrict__ token_of_row, float* __restrict__ row_weight,
                            int* __restrict__ row_of) {
    int idx = blockIdx.x * blockDim.x + threadIdx.x;
    if (idx >= T_TOK * TOPK) return;
    int t = idx >> 1;
    int e = topi[idx];
    int pos = offs[e] + atomicAdd(&cursors[e], 1);
    token_of_row[pos] = t;
    row_weight[pos] = topw[idx];
    row_of[idx] = pos;
}

// ---------------- 5: x fp32 -> bf16, 4-wide (+ one zero row for padded slots) ----------------
__global__ void moe_cvt_x(const float* __restrict__ x, bf16_t* __restrict__ xb) {
    int i4 = blockIdx.x * blockDim.x + threadIdx.x;     // element-quad index
    if (i4 >= (T_TOK + 1) * HID / 4) return;
    v4bf o;
    if (i4 < T_TOK * HID / 4) {
        v4f f = *(const v4f*)(x + (size_t)i4 * 4);
#pragma unroll
        for (int j = 0; j < 4; ++j) o[j] = (bf16_t)f[j];
    } else {
#pragma unroll
        for (int j = 0; j < 4; ++j) o[j] = (bf16_t)0.f;
    }
    *(v4bf*)(xb + (size_t)i4 * 4) = o;
}

// ---------------- 6: GEMM1  act = silu(x@w13_gate) * (x@w13_up)  [bf16 WMMA] ----------------
__global__ __launch_bounds__(256, 1)
void moe_gemm1(const bf16_t* __restrict__ xb, const float* __restrict__ w13,
               const int* __restrict__ token_of_row, const int* __restrict__ tile_expert,
               bf16_t* __restrict__ act) {
    const int wave = threadIdx.x >> 5;
    const int lane = threadIdx.x & 31;
    const int lo = lane & 15, hi = lane >> 4;
    const int row0 = blockIdx.y * 16;
    const int col0 = blockIdx.x * 128 + wave * 16;      // intermediate-channel tile base
    const int e = tile_expert[blockIdx.y];
    const int t = token_of_row[row0 + lo];
    const bf16_t* arow = xb + (size_t)((t < 0) ? T_TOK : t) * HID;   // zero row for padding
    const int n = col0 + lo;
    const float* wg = w13 + (size_t)e * (2 * INTER) * HID + (size_t)n * HID + hi * 8;
    const float* wu = wg + (size_t)INTER * HID;

    v8f cg = {}; v8f cu = {};
    for (int kb = 0; kb < HID; kb += 32) {
        v16bf a  = load_a_bf16(arow + kb, hi);
        v16bf bg = load_b_f32(wg + kb);
        v16bf bu = load_b_f32(wu + kb);
        cg = __builtin_amdgcn_wmma_f32_16x16x32_bf16(false, a, false, bg, (short)0, cg, false, false);
        cu = __builtin_amdgcn_wmma_f32_16x16x32_bf16(false, a, false, bu, (short)0, cu, false, false);
    }
    // C/D layout: col = lane&15, rows = (lane>>4)*8 + v. Fused SwiGLU epilogue.
#pragma unroll
    for (int v = 0; v < 8; ++v) {
        float g = cg[v], u = cu[v];
        act[(size_t)(row0 + hi * 8 + v) * INTER + n] = (bf16_t)(fast_silu(g) * u);
    }
}

// ---------------- 7: GEMM2  yp = (act @ w2^T) * router_weight ----------------
__global__ __launch_bounds__(256, 1)
void moe_gemm2(const bf16_t* __restrict__ act, const float* __restrict__ w2,
               const int* __restrict__ tile_expert, const float* __restrict__ row_weight,
               float* __restrict__ yp) {
    const int wave = threadIdx.x >> 5;
    const int lane = threadIdx.x & 31;
    const int lo = lane & 15, hi = lane >> 4;
    const int row0 = blockIdx.y * 16;
    const int col0 = blockIdx.x * 128 + wave * 16;      // hidden-channel tile base
    const int e = tile_expert[blockIdx.y];
    const bf16_t* arow = act + (size_t)(row0 + lo) * INTER;          // padded rows are zero
    const int n = col0 + lo;
    const float* wrow = w2 + (size_t)e * HID * INTER + (size_t)n * INTER + hi * 8;

    v8f c = {};
    for (int kb = 0; kb < INTER; kb += 32) {
        v16bf a = load_a_bf16(arow + kb, hi);
        v16bf b = load_b_f32(wrow + kb);
        c = __builtin_amdgcn_wmma_f32_16x16x32_bf16(false, a, false, b, (short)0, c, false, false);
    }
#pragma unroll
    for (int v = 0; v < 8; ++v) {
        int m = row0 + hi * 8 + v;
        yp[(size_t)m * HID + n] = c[v] * row_weight[m];
    }
}

// ---------------- 8: combine the two expert rows per token (4-wide) ----------------
__global__ void moe_combine(const float* __restrict__ yp, const int* __restrict__ row_of,
                            float* __restrict__ out) {
    int i4 = blockIdx.x * blockDim.x + threadIdx.x;     // quad index over [T, H]
    if (i4 >= T_TOK * HID / 4) return;
    int t = i4 / (HID / 4);
    int h4 = i4 - t * (HID / 4);
    int r0 = row_of[2 * t], r1 = row_of[2 * t + 1];
    v4f a = *(const v4f*)(yp + (size_t)r0 * HID + h4 * 4);
    v4f b = *(const v4f*)(yp + (size_t)r1 * HID + h4 * 4);
    *(v4f*)(out + (size_t)i4 * 4) = a + b;
}

// ---------------- launch ----------------
static inline size_t align_up(size_t v) { return (v + 255) & ~(size_t)255; }

extern "C" void kernel_launch(void* const* d_in, const int* in_sizes, int n_in,
                              void* d_out, int out_size, void* d_ws, size_t ws_size,
                              hipStream_t stream) {
    (void)in_sizes; (void)n_in; (void)out_size; (void)ws_size;
    const float* x      = (const float*)d_in[0];
    const float* logits = (const float*)d_in[1];
    const float* w13    = (const float*)d_in[2];
    const float* w2     = (const float*)d_in[3];
    float* out = (float*)d_out;

    // workspace carve-out (~27 MB)
    char* p = (char*)d_ws;
    bf16_t* xb           = (bf16_t*)p; p += align_up((size_t)(T_TOK + 1) * HID * sizeof(bf16_t));
    bf16_t* act          = (bf16_t*)p; p += align_up((size_t)SROWS * INTER * sizeof(bf16_t));
    float*  yp           = (float*)p;  p += align_up((size_t)SROWS * HID * sizeof(float));
    int*    counts       = (int*)p;    p += align_up(NEXP * sizeof(int));
    int*    cursors      = (int*)p;    p += align_up(NEXP * sizeof(int));
    int*    offs         = (int*)p;    p += align_up((NEXP + 1) * sizeof(int));
    int*    topi         = (int*)p;    p += align_up((size_t)T_TOK * TOPK * sizeof(int));
    float*  topw         = (float*)p;  p += align_up((size_t)T_TOK * TOPK * sizeof(float));
    int*    token_of_row = (int*)p;    p += align_up(SROWS * sizeof(int));
    float*  row_weight   = (float*)p;  p += align_up(SROWS * sizeof(float));
    int*    row_of       = (int*)p;    p += align_up((size_t)T_TOK * TOPK * sizeof(int));
    int*    tile_expert  = (int*)p;    p += align_up(MTILES * sizeof(int));

    moe_init<<<(SROWS + 255) / 256, 256, 0, stream>>>(counts, cursors, token_of_row, row_weight);
    moe_route<<<(T_TOK + 255) / 256, 256, 0, stream>>>(logits, counts, topi, topw);
    moe_offsets<<<1, 256, 0, stream>>>(counts, offs, tile_expert);
    moe_scatter<<<(T_TOK * TOPK + 255) / 256, 256, 0, stream>>>(topi, topw, offs, cursors,
                                                                token_of_row, row_weight, row_of);
    moe_cvt_x<<<((T_TOK + 1) * HID / 4 + 255) / 256, 256, 0, stream>>>(x, xb);
    moe_gemm1<<<dim3(INTER / 128, MTILES), 256, 0, stream>>>(xb, w13, token_of_row, tile_expert, act);
    moe_gemm2<<<dim3(HID / 128, MTILES), 256, 0, stream>>>(act, w2, tile_expert, row_weight, yp);
    moe_combine<<<(T_TOK * HID / 4 + 255) / 256, 256, 0, stream>>>(yp, row_of, out);
}
